// GRU_46308337385722
// MI455X (gfx1250) — matrix operand order repaired and checked by reading, not compiled, not run on hardware
//
#include <hip/hip_runtime.h>
#include <hip/hip_bf16.h>
#include <math.h>

// ---------------------------------------------------------------------------
// GRU on MI455X (gfx1250, wave32, WMMA + TDM).
//   B=64, T=512, I=H=1024, 3H=3072.
//   Phase A: convert to bf16 (x, Wx^T, Wh^T), zero h ping-pong + sync cells.
//   Phase B: xt[B*T,3H] = x @ Wx^T + bx (+bh for r,u gates)  -- bf16 WMMA GEMM
//   Phase C: persistent scan kernel, 64 workgroups; Wh slice staged into LDS
//            by ONE tensor_load_to_lds (TDM, 3-D descriptor, 96 KB);
//            v_wmma_f32_16x16x32_bf16 recurrent GEMM + fp32 gates;
//            global_prefetch of next step's xt; device-wide barrier per step.
// ---------------------------------------------------------------------------

typedef __attribute__((ext_vector_type(16))) __bf16 v16bf;
typedef __attribute__((ext_vector_type(8)))  float  v8f;
typedef __attribute__((ext_vector_type(4)))  unsigned int v4u32;
typedef __attribute__((ext_vector_type(8)))  int    v8i32;
typedef __attribute__((ext_vector_type(4)))  int    v4i32;

#define WMMA_BF16(a, b, c) \
    __builtin_amdgcn_wmma_f32_16x16x32_bf16(false, (a), false, (b), (short)0, (c), false, false)

__device__ __forceinline__ unsigned short f2bf(float f) {
    union { float f; unsigned u; } v; v.f = f;
    unsigned r = v.u + 0x7FFFu + ((v.u >> 16) & 1u);   // round-to-nearest-even
    return (unsigned short)(r >> 16);
}

__device__ __forceinline__ v16bf load_frag_2x16B(const unsigned short* p0,
                                                 const unsigned short* p1) {
    v16bf a;
    *reinterpret_cast<uint4*>(&a)       = *reinterpret_cast<const uint4*>(p0);
    *(reinterpret_cast<uint4*>(&a) + 1) = *reinterpret_cast<const uint4*>(p1);
    return a;
}

// ------------------------------- Phase A -----------------------------------

__global__ void k_zero_u32(unsigned int* __restrict__ p, int n) {
    int i = blockIdx.x * blockDim.x + threadIdx.x;
    if (i < n) p[i] = 0u;
}

__global__ void k_cvt_bf16(const float* __restrict__ src,
                           unsigned short* __restrict__ dst, int n) {
    int stride = gridDim.x * blockDim.x;
    for (int i = blockIdx.x * blockDim.x + threadIdx.x; i < n; i += stride)
        dst[i] = f2bf(src[i]);
}

// src: [3072][1024] row-major -> dst: [1024][3072] (transposed) bf16
__global__ void k_cvtT_bf16(const float* __restrict__ src,
                            unsigned short* __restrict__ dst) {
    const int n = 1024 * 3072;
    int stride = gridDim.x * blockDim.x;
    for (int i = blockIdx.x * blockDim.x + threadIdx.x; i < n; i += stride) {
        int k = i / 3072;          // 0..1023
        int c = i - k * 3072;      // 0..3071
        dst[i] = f2bf(src[(size_t)c * 1024 + k]);
    }
}

// ------------------------------- Phase B -----------------------------------
// xt = x @ Wx^T + bx (+ bh for gate columns < 2048).
// grid(48, 256), block 256 (8 waves). Each wave: 16 rows x 64 cols, K=1024.

__global__ void k_xproj(const unsigned short* __restrict__ xbf,   // [32768][1024]
                        const unsigned short* __restrict__ WxT,   // [1024][3072]
                        const float* __restrict__ bx,
                        const float* __restrict__ bh,
                        float* __restrict__ xt) {                 // [32768][3072]
    const int lane  = threadIdx.x & 31;
    const int wave  = threadIdx.x >> 5;
    const int m0    = (blockIdx.y * 8 + wave) * 16;
    const int n0    = blockIdx.x * 64;
    const int arow  = m0 + (lane & 15);
    const int khalf = (lane >> 4) * 8;

    v8f acc[4] = {v8f{}, v8f{}, v8f{}, v8f{}};

    for (int kc = 0; kc < 1024; kc += 32) {
        // A fragment (ISA 16-bit A 16x32 layout): K {khalf..khalf+7, 16+khalf..}
        const unsigned short* ap = xbf + (size_t)arow * 1024 + kc + khalf;
        v16bf a = load_frag_2x16B(ap, ap + 16);
        // B fragments: lane = N column, K {khalf..khalf+7, 16+khalf..} (same as A)
        const unsigned short* bp = WxT + (size_t)kc * 3072 + n0 + (lane & 15);
#pragma unroll
        for (int s = 0; s < 4; ++s) {
            unsigned short bt[16];
#pragma unroll
            for (int e = 0; e < 16; ++e) {
                int kk = (e < 8) ? (khalf + e) : (16 + khalf + e - 8);
                bt[e] = bp[(size_t)kk * 3072 + s * 16];
            }
            v16bf b = __builtin_bit_cast(v16bf, bt);
            acc[s] = WMMA_BF16(a, b, acc[s]);
        }
    }

    const int nlo  = lane & 15;
    const int mofs = (lane >> 4) * 8;     // C/D layout: lanes 16-31 hold M=r+8
#pragma unroll
    for (int s = 0; s < 4; ++s) {
        int g = n0 + s * 16 + nlo;
        float bias = bx[g] + (g < 2048 ? bh[g] : 0.0f);
#pragma unroll
        for (int r = 0; r < 8; ++r) {
            int m = m0 + mofs + r;
            xt[(size_t)m * 3072 + g] = acc[s][r] + bias;
        }
    }
}

// ------------------------------- Phase C -----------------------------------
// Persistent scan: 64 workgroups x 128 threads (4 waves). Workgroup g owns
// h columns [g*16, g*16+16). Wh slice (3 gates x 1024 K x 16 N bf16 = 96 KB)
// staged into LDS as [gate][k][n] by a single TDM tensor_load_to_lds with a
// 3-D descriptor: tile(16 x 1024 x 3), dim0_stride=3072, dim1_stride=1024.

__global__ void k_gru_scan(const unsigned short* __restrict__ WhT,  // [1024][3072]
                           const float* __restrict__ xt,            // [32768][3072]
                           const float* __restrict__ bh,            // [3072]
                           unsigned short* __restrict__ hbuf,       // [2][64*1024] bf16
                           float* __restrict__ out,                 // [64][512][1024]
                           unsigned int* __restrict__ syncbuf) {
    extern __shared__ unsigned short ldsB[];   // 3*1024*16 bf16 = 96 KB

    const int tid  = threadIdx.x;
    const int lane = tid & 31;
    const int wave = tid >> 5;
    const int col0 = blockIdx.x * 16;

    // ---- TDM stage of Wh slice (once; reused for all 512 steps) ----
    if (tid < 32) {
        unsigned long long ga = (unsigned long long)(const void*)(WhT + col0);
        unsigned lds_addr = (unsigned)(unsigned long long)(const void*)ldsB;

        v4u32 g0;
        g0[0] = 1u;                         // count=1 (valid), user mode
        g0[1] = lds_addr;                   // LDS byte address
        g0[2] = (unsigned)ga;               // global_addr[31:0]
        g0[3] = (unsigned)((ga >> 32) & 0x01FFFFFFu) | 0x80000000u; // [56:32] | type=2

        v8i32 g1;
        g1[0] = (int)(1u << 16);            // wg_mask=0, data_size=1 (2 bytes)
        g1[1] = (int)(3072u << 16);         // tensor_dim0[15:0]=3072
        g1[2] = (int)(1024u << 16);         // tensor_dim0[31:16]=0 | tensor_dim1[15:0]=1024
        g1[3] = (int)(16u << 16);           // tensor_dim1[31:16]=0 | tile_dim0=16
        g1[4] = (int)(1024u | (3u << 16));  // tile_dim1=1024 | tile_dim2=3 (gates)
        g1[5] = 3072;                       // tensor_dim0_stride lo (elements)
        g1[6] = (int)(1024u << 16);         // stride0 hi=0 | tensor_dim1_stride[15:0]=1024
        g1[7] = 0;                          // tensor_dim1_stride hi

        v4i32 g2;
        g2[0] = 3;                          // tensor_dim2 = 3 (gates)
        g2[1] = 0;                          // tensor_dim3 (unused)
        g2[2] = 0;                          // tensor_dim2_stride lo (unused: tile_dim3=0)
        g2[3] = 0;                          // stride hi | tile_dim3=0

        v4i32 g3 = {0, 0, 0, 0};            // no dim4
        v8i32 g4 = {0, 0, 0, 0, 0, 0, 0, 0}; // extra group (clang-23 6-arg form)

        __builtin_amdgcn_tensor_load_to_lds(g0, g1, g2, g3, g4, 0);
        __builtin_amdgcn_s_wait_tensorcnt(0);
    }
    __syncthreads();

    const int m0    = wave * 16;              // 4 waves cover B=64 rows
    const int arow  = m0 + (lane & 15);
    const int khalf = (lane >> 4) * 8;
    const int nlo   = lane & 15;
    const int mofs  = (lane >> 4) * 8;
    const int n     = col0 + nlo;
    const float bhn = bh[2048 + n];

    unsigned int* counter = syncbuf;
    unsigned int* genflag = syncbuf + 16;

    for (int t = 0; t < 512; ++t) {
        const unsigned short* hb  = hbuf + (size_t)(t & 1) * 65536;
        unsigned short*       hbn = hbuf + (size_t)((t + 1) & 1) * 65536;

        v8f accr = {}, accu = {}, accn = {};
        for (int kc = 0; kc < 1024; kc += 32) {
            const unsigned short* ap = hb + (size_t)arow * 1024 + kc + khalf;
            v16bf a = load_frag_2x16B(ap, ap + 16);
            const unsigned short* lb = ldsB + kc * 16 + nlo;
            unsigned short tr[16], tu[16], tn[16];
#pragma unroll
            for (int e = 0; e < 16; ++e) {
                int kk = (e < 8) ? (khalf + e) : (16 + khalf + e - 8);
                tr[e] = lb[kk * 16];
                tu[e] = lb[16384 + kk * 16];
                tn[e] = lb[32768 + kk * 16];
            }
            v16bf br = __builtin_bit_cast(v16bf, tr);
            v16bf bu = __builtin_bit_cast(v16bf, tu);
            v16bf bn = __builtin_bit_cast(v16bf, tn);
            accr = WMMA_BF16(a, br, accr);
            accu = WMMA_BF16(a, bu, accu);
            accn = WMMA_BF16(a, bn, accn);
        }

        // Gates + state update (fp32). h_prev fp32 comes from out[t-1].
#pragma unroll
        for (int r = 0; r < 8; ++r) {
            int b = m0 + mofs + r;
            size_t xtb = ((size_t)b * 512 + t) * 3072;
            float xr = xt[xtb + n];
            float xu = xt[xtb + 1024 + n];
            float xn = xt[xtb + 2048 + n];
            float hp = (t == 0) ? 0.0f
                                : out[((size_t)b * 512 + (t - 1)) * 1024 + n];
            float rg = 1.0f / (1.0f + __expf(-(accr[r] + xr)));
            float zg = 1.0f / (1.0f + __expf(-(accu[r] + xu)));
            float ng = tanhf(xn + rg * (accn[r] + bhn));
            float hn = zg * hp + (1.0f - zg) * ng;
            out[((size_t)b * 512 + t) * 1024 + n] = hn;
            hbn[b * 1024 + n] = f2bf(hn);

            // Hide HBM latency: prefetch next step's xt operands (streamed buf)
            if (t < 511) {
                const float* pf = xt + xtb + 3072;   // (b, t+1) row
                __builtin_prefetch(pf + n, 0, 1);
                __builtin_prefetch(pf + 1024 + n, 0, 1);
                __builtin_prefetch(pf + 2048 + n, 0, 1);
            }
        }

        // Device-wide barrier (monotonic generation: no reset races)
        __syncthreads();
        if (tid == 0) {
            __threadfence();
            unsigned int target  = (unsigned int)(t + 1);
            unsigned int arrived = atomicAdd(counter, 1u);
            if (arrived == target * 64u - 1u) {
                atomicExch(genflag, target);
            } else {
                volatile unsigned int* vg = genflag;
                while (*vg < target) __builtin_amdgcn_s_sleep(1);
            }
        }
        __syncthreads();
        __threadfence();   // acquire published h before next step
    }
}

// ------------------------------- Launch ------------------------------------

extern "C" void kernel_launch(void* const* d_in, const int* in_sizes, int n_in,
                              void* d_out, int out_size, void* d_ws, size_t ws_size,
                              hipStream_t stream) {
    const float* x  = (const float*)d_in[0];   // [64,512,1024]
    const float* Wx = (const float*)d_in[1];   // [3072,1024]
    const float* bx = (const float*)d_in[2];   // [3072]
    const float* Wh = (const float*)d_in[3];   // [3072,1024]
    const float* bh = (const float*)d_in[4];   // [3072]
    float* out = (float*)d_out;                // [64,512,1024]

    char* ws = (char*)d_ws;
    size_t off = 0;
    float* xt            = (float*)(ws + off);          off += (size_t)32768 * 3072 * 4;
    unsigned short* xbf  = (unsigned short*)(ws + off); off += (size_t)33554432 * 2;
    unsigned short* WxT  = (unsigned short*)(ws + off); off += (size_t)1024 * 3072 * 2;
    unsigned short* WhT  = (unsigned short*)(ws + off); off += (size_t)1024 * 3072 * 2;
    unsigned short* hbuf = (unsigned short*)(ws + off); off += (size_t)2 * 64 * 1024 * 2;
    unsigned int* syncb  = (unsigned int*)(ws + off);   off += 256;
    (void)ws_size; (void)in_sizes; (void)n_in; (void)out_size;

    // Phase A: zero hbuf + sync (contiguous: 65536 + 64 u32 words), convert to bf16
    k_zero_u32<<<(65600 + 255) / 256, 256, 0, stream>>>((unsigned int*)hbuf, 65600);
    k_cvt_bf16<<<4096, 256, 0, stream>>>(x, xbf, 33554432);
    k_cvtT_bf16<<<2048, 256, 0, stream>>>(Wx, WxT);
    k_cvtT_bf16<<<2048, 256, 0, stream>>>(Wh, WhT);

    // Phase B: input projection GEMM (2048 M-tiles x 192 N-tiles of 16x16)
    dim3 gB(48, 256);
    k_xproj<<<gB, 256, 0, stream>>>(xbf, WxT, bx, bh, xt);

    // Phase C: persistent recurrent scan, 96 KB dynamic LDS per workgroup
    k_gru_scan<<<64, 128, 3 * 1024 * 16 * 2, stream>>>(WhT, xt, bh, hbuf, out, syncb);
}
